// Linear_71262097375897
// MI455X (gfx1250) — matrix operand order repaired and checked by reading, not compiled, not run on hardware
//
#include <hip/hip_runtime.h>
#include <hip/hip_bf16.h>

// Q4_0 dequant-GEMM for MI455X (gfx1250, wave32, WMMA).
//   y[16, 11008] = x[16, 4096] @ W^T + bias,  W[o,k] = s[o,k/32]*(q[o,k]-8)
//
// Memory-bound: qweight (int32 per 4-bit value) ~180 MB; AI ~ 8 FLOP/B.
// To actually reach bandwidth we need memory-level parallelism:
//   - one 256-thread block (8 waves) per 16x16 N-tile
//   - 8-way K-split: wave w handles k-blocks [w*16, w*16+16)
//   - 688 blocks x 8 waves = 5504 waves, ~22 MB of loads in flight
//   - partial 16x16 f32 tiles reduced through LDS (fixed order -> deterministic)
// Quant block (32) == WMMA K, so the scale folds into the f16 B fragment.

#define OUT_F 11008
#define IN_F  4096
#define NB    128   // IN_F / 32 quant blocks
#define BLK   32
#define KSPLIT 8                  // waves per block == K-split ways
#define KB_PER_WAVE (NB / KSPLIT) // 16 quant blocks per wave

typedef __attribute__((ext_vector_type(16))) _Float16 v16h;
typedef __attribute__((ext_vector_type(8)))  float    v8f;

__global__ __launch_bounds__(32 * KSPLIT)
void q4_linear_wmma(const float* __restrict__ x,       // [16][IN_F]
                    const int*   __restrict__ qw,      // [OUT_F][NB][BLK]
                    const float* __restrict__ scales,  // [OUT_F][NB]
                    const float* __restrict__ bias,    // [OUT_F]
                    float*       __restrict__ y)       // [16][OUT_F]
{
    __shared__ float red[KSPLIT * 256];   // 8 partial 16x16 f32 tiles (8 KB)

    const int lane = threadIdx.x & 31;
    const int w    = threadIdx.x >> 5;    // wave id == K-split slice
    const int o0   = blockIdx.x * 16;     // N-tile origin (688 tiles)

    const int h  = lane >> 4;   // half-wave select
    const int mn = lane & 15;   // A row / B column

    const int o = o0 + mn;      // this lane's output feature (B column)

    const float* xrow = x + (size_t)mn * IN_F;
    const int*   qrow = qw + (size_t)o * (NB * BLK);
    const float* srow = scales + (size_t)o * NB;

    v8f acc = {};

    const int kb0 = w * KB_PER_WAVE;
#pragma unroll 2
    for (int i = 0; i < KB_PER_WAVE; ++i) {
        const int kb    = kb0 + i;
        const int kbase = kb * BLK;

        // ---- A fragment: x in f16.  ISA layout (16-bit A 16x32):
        //   halves 0..7  -> K = 8*h + e ; halves 8..15 -> K = 16 + 8*h + (e-8)
        const float* xp0 = xrow + kbase + h * 8;
        const float* xp1 = xp0 + 16;
        v16h a;
#pragma unroll
        for (int e = 0; e < 8; ++e) {
            a[e]     = (_Float16)xp0[e];
            a[e + 8] = (_Float16)xp1[e];
        }

        // ---- B fragment: dequantized W column, same K order as A:
        //   e < 8 -> K = 8*h + e ; e >= 8 -> K = 16 + 8*h + (e-8)
        const float s  = srow[kb];
        const float s8 = s * 8.0f;
        const int*  qp = qrow + kbase + h * 8;
        v16h b;
#pragma unroll
        for (int e = 0; e < 8; ++e) {
            // (q-8)*s == fma(q, s, -8s): cvt_i32_f32 + pk_fma + cvt_pk_f16
            b[e]     = (_Float16)__builtin_fmaf((float)qp[e],     s, -s8);
            b[e + 8] = (_Float16)__builtin_fmaf((float)qp[e + 16], s, -s8);
        }

        acc = __builtin_amdgcn_wmma_f32_16x16x32_f16(
            /*neg_a=*/false, a, /*neg_b=*/false, b,
            /*c_mod=*/(short)0, acc, /*reuse_a=*/false, /*reuse_b=*/false);
    }

    // ---- Stage partial tile: slot = (r, lane), one tile per wave.
#pragma unroll
    for (int r = 0; r < 8; ++r)
        red[w * 256 + r * 32 + lane] = acc[r];

    __syncthreads();

    // ---- Cross-wave reduction: 256 threads, each owns one (r, lane) slot.
    // Fixed summation order -> bitwise deterministic.
    const int lane2 = threadIdx.x & 31;
    const int r2    = threadIdx.x >> 5;          // 0..7
    float sum = 0.0f;
#pragma unroll
    for (int ww = 0; ww < KSPLIT; ++ww)
        sum += red[ww * 256 + r2 * 32 + lane2];  // conflict-free: stride-1 in lane

    // D layout: slot (r, lane) -> M = r + 8*(lane>>4), N = lane&15
    const int n2 = lane2 & 15;
    const int m2 = r2 + 8 * (lane2 >> 4);
    sum += bias[o0 + n2];
    y[(size_t)m2 * OUT_F + o0 + n2] = sum;
}

extern "C" void kernel_launch(void* const* d_in, const int* in_sizes, int n_in,
                              void* d_out, int out_size, void* d_ws, size_t ws_size,
                              hipStream_t stream) {
    (void)in_sizes; (void)n_in; (void)out_size; (void)d_ws; (void)ws_size;
    const float* x  = (const float*)d_in[0];
    const int*   qw = (const int*)  d_in[1];
    const float* sc = (const float*)d_in[2];
    const float* bs = (const float*)d_in[3];
    float*       y  = (float*)d_out;

    const int n_tiles = OUT_F / 16;   // 688 blocks, 8 waves each
    q4_linear_wmma<<<n_tiles, 32 * KSPLIT, 0, stream>>>(x, qw, sc, bs, y);
}
